// sju_53214644798203
// MI455X (gfx1250) — matrix operand order repaired and checked by reading, not compiled, not run on hardware
//
#include <hip/hip_runtime.h>

typedef __bf16 bf16;
typedef __attribute__((ext_vector_type(8)))  __bf16 v8bf;
typedef __attribute__((ext_vector_type(16))) __bf16 v16bf;
typedef __attribute__((ext_vector_type(8)))  float  v8f;

#define NHID   512
#define LJ     17
#define NHEAD  8
#define DK     64
#define DFF    2048
#define NBATCH 256
#define M_X    (NBATCH * LJ)      // 4352 tokens from x
#define M_AX   (NBATCH * 2 * LJ)  // 8704 tokens from ax

#ifdef __has_builtin
#if __has_builtin(__builtin_amdgcn_global_load_async_to_lds_b128)
#define HAVE_ASYNC_LDS 1
#endif
#endif

// ---- skeleton graph tables (0-indexed, pad never dereferenced) ----
__constant__ int c_neigh[LJ][5] = {
  {0,1,2,5,6},{0,1,3,17,17},{0,2,4,17,17},{1,3,17,17,17},{2,4,17,17,17},
  {0,5,7,11,17},{0,6,8,12,17},{5,7,9,17,17},{6,8,12,17,17},{7,9,17,17,17},
  {8,10,17,17,17},{8,11,13,17,17},{10,12,14,17,17},{11,13,15,17,17},
  {12,14,16,17,17},{13,15,17,17,17},{14,16,17,17,17}};
__constant__ int c_slen[LJ] = {4,2,2,1,1,3,3,2,2,1,1,2,2,2,2,1,1};

// ===================== conversion kernels =====================
__global__ void cvt_f32_bf16(const float* __restrict__ src, bf16* __restrict__ dst, int n) {
  int i = blockIdx.x * blockDim.x + threadIdx.x;
  if (i < n) dst[i] = (bf16)src[i];
}

// x: [B,512,17] -> X: [B*17, 512] bf16
__global__ void cvt_x_kernel(const float* __restrict__ x, bf16* __restrict__ X) {
  int bl = blockIdx.x; int b = bl / LJ, l = bl % LJ;
  for (int c = threadIdx.x; c < NHID; c += blockDim.x)
    X[(size_t)bl * NHID + c] = (bf16)x[(size_t)b * NHID * LJ + (size_t)c * LJ + l];
}

// ax: [B,512,2,17] -> XA: [B*34, 512] bf16 with row m = b*34 + a*17 + l
__global__ void cvt_ax_kernel(const float* __restrict__ ax, bf16* __restrict__ XA) {
  int m = blockIdx.x; int b = m / (2 * LJ); int r = m % (2 * LJ);
  for (int c = threadIdx.x; c < NHID; c += blockDim.x)
    XA[(size_t)m * NHID + c] = (bf16)ax[(size_t)b * NHID * 2 * LJ + (size_t)c * 2 * LJ + r];
}

__global__ void concat_bias_kernel(const float* __restrict__ qb, const float* __restrict__ kb,
                                   const float* __restrict__ vb, float* __restrict__ dst) {
  int i = blockIdx.x * blockDim.x + threadIdx.x;
  if (i < 512)        dst[i] = qb[i];
  else if (i < 1024)  dst[i] = kb[i - 512];
  else if (i < 1536)  dst[i] = vb[i - 1024];
}

// ===================== WMMA bf16 GEMM: D[M,N] = A[M,K] * Bw[N,K]^T =====================
// Block tile 128x128, 8 waves (4 along M x 2 along N), wave tile 32x64 => 2x4 wmma tiles.
// Double-buffered LDS fed by GLOBAL_LOAD_ASYNC_TO_LDS_B128 (ASYNCcnt) when available.
#define BM 128
#define BN 128
#define BK 32
#define LDSS 40   // padded half-element row stride: 80B => 20-bank step, conflict-free, 16B aligned

#if defined(HAVE_ASYNC_LDS)
typedef int v4i __attribute__((__vector_size__(16)));
typedef __attribute__((address_space(1))) v4i* as1_v4i_ptr;
typedef __attribute__((address_space(3))) v4i* as3_v4i_ptr;

__device__ __forceinline__ void async_copy16(const bf16* g, bf16* l) {
  __builtin_amdgcn_global_load_async_to_lds_b128(
      (as1_v4i_ptr)(g), (as3_v4i_ptr)(l), 0, 0);
}
__device__ __forceinline__ void wait_async0() {
#if __has_builtin(__builtin_amdgcn_s_wait_asynccnt)
  __builtin_amdgcn_s_wait_asynccnt(0);
#else
  asm volatile("s_wait_asynccnt 0" ::: "memory");
#endif
}
#endif

// MODE 0: f32 out, +bias | MODE 1: bf16 out, relu(x+bias) | MODE 2: f32 out, raw (bias unused)
template <int MODE>
__global__ __launch_bounds__(256) void gemm_bf16_wmma(
    const bf16* __restrict__ A, const bf16* __restrict__ Bw,
    const float* __restrict__ bias, void* __restrict__ Cout,
    int M, int N, int K)
{
  __shared__ bf16 As[2][BM * LDSS];
  __shared__ bf16 Bs[2][BM * LDSS];

  const int tid  = threadIdx.x;
  const int lane = tid & 31;
  const int wave = tid >> 5;
  const int wm   = wave & 3;        // 4 waves along M (32 rows each)
  const int wn   = wave >> 2;       // 2 waves along N (64 cols each)
  const int bm0  = blockIdx.y * BM;
  const int bn0  = blockIdx.x * BN;
  const int lrow    = lane & 15;
  const int halfsel = lane >> 4;    // 0: lanes 0-15, 1: lanes 16-31

  // per-thread global->LDS staging coordinates (128x32 halves, 16 halves/thread)
  const int sidx = tid * 16;
  const int srow = sidx >> 5;       // 0..127
  const int scol = sidx & 31;       // 0 or 16
  const bf16* ga = A  + (size_t)(bm0 + srow) * K + scol;
  const bf16* gb = Bw + (size_t)(bn0 + srow) * K + scol;
  const int soff = srow * LDSS + scol;

  v8f acc[2][4] = {};
  const int nk = K / BK;

#if defined(HAVE_ASYNC_LDS)
  // ---- async double-buffered pipeline: one barrier + one asynccnt wait per K-step ----
  async_copy16(ga,     &As[0][soff]);
  async_copy16(ga + 8, &As[0][soff + 8]);
  async_copy16(gb,     &Bs[0][soff]);
  async_copy16(gb + 8, &Bs[0][soff + 8]);
  for (int kt = 0; kt < nk; ++kt) {
    const int cur = kt & 1;
    wait_async0();                 // this wave's tile-cur writes landed in LDS
    __syncthreads();               // everyone's tile-cur visible; prior reads of cur^1 done
    if (kt + 1 < nk) {             // stream next tile into the other buffer during compute
      const bf16* pa = ga + (kt + 1) * BK;
      const bf16* pb = gb + (kt + 1) * BK;
      async_copy16(pa,     &As[cur ^ 1][soff]);
      async_copy16(pa + 8, &As[cur ^ 1][soff + 8]);
      async_copy16(pb,     &Bs[cur ^ 1][soff]);
      async_copy16(pb + 8, &Bs[cur ^ 1][soff + 8]);
    }
#else
  for (int kt = 0; kt < nk; ++kt) {
    const int cur = 0;
    { // de-serialized staging: issue all 4 loads, then all 4 stores
      const bf16* pa = ga + kt * BK;
      const bf16* pb = gb + kt * BK;
      v8bf t0 = *(const v8bf*)(pa);
      v8bf t1 = *(const v8bf*)(pa + 8);
      v8bf t2 = *(const v8bf*)(pb);
      v8bf t3 = *(const v8bf*)(pb + 8);
      *(v8bf*)&As[0][soff]     = t0;
      *(v8bf*)&As[0][soff + 8] = t1;
      *(v8bf*)&Bs[0][soff]     = t2;
      *(v8bf*)&Bs[0][soff + 8] = t3;
      if (kt + 1 < nk) {
        __builtin_prefetch(pa + BK, 0, 3);
        __builtin_prefetch(pb + BK, 0, 3);
      }
    }
    __syncthreads();
#endif

    v16bf afrag[2], bfrag[4];
#pragma unroll
    for (int mi = 0; mi < 2; ++mi) {
      // ISA 16-bit A 16x32 layout: lanes0-15 K{0-7,16-23}, lanes16-31 K{8-15,24-31}
      const int row = wm * 32 + mi * 16 + lrow;
      v8bf lo = *(const v8bf*)&As[cur][row * LDSS + halfsel * 8];
      v8bf hi = *(const v8bf*)&As[cur][row * LDSS + halfsel * 8 + 16];
      afrag[mi] = __builtin_shufflevector(lo, hi, 0,1,2,3,4,5,6,7,8,9,10,11,12,13,14,15);
    }
#pragma unroll
    for (int ni = 0; ni < 4; ++ni) {
      // B 32x16: lane=column; lanes0-15 K{0-7,16-23}, lanes16-31 K{8-15,24-31}
      const int col = wn * 64 + ni * 16 + lrow;
      v8bf lo = *(const v8bf*)&Bs[cur][col * LDSS + halfsel * 8];
      v8bf hi = *(const v8bf*)&Bs[cur][col * LDSS + halfsel * 8 + 16];
      bfrag[ni] = __builtin_shufflevector(lo, hi, 0,1,2,3,4,5,6,7,8,9,10,11,12,13,14,15);
    }
#pragma unroll
    for (int mi = 0; mi < 2; ++mi)
#pragma unroll
      for (int ni = 0; ni < 4; ++ni)
        acc[mi][ni] = __builtin_amdgcn_wmma_f32_16x16x32_bf16(
            false, afrag[mi], false, bfrag[ni], (short)0, acc[mi][ni], false, false);
#if !defined(HAVE_ASYNC_LDS)
    __syncthreads();               // single-buffer: protect tile before next overwrite
#endif
  }

  // C/D layout: VGPR e -> M = e + 8*halfsel, N = lane&15. Branch-free templated epilogue.
#pragma unroll
  for (int ni = 0; ni < 4; ++ni) {
    const int n = bn0 + wn * 64 + ni * 16 + lrow;
    const float bv = (MODE == 2) ? 0.0f : bias[n];
#pragma unroll
    for (int mi = 0; mi < 2; ++mi) {
      const int m0 = bm0 + wm * 32 + mi * 16 + halfsel * 8;
      if (MODE == 1) {
        bf16* p = (bf16*)Cout + (size_t)m0 * N + n;
#pragma unroll
        for (int e = 0; e < 8; ++e) {
          float v = acc[mi][ni][e] + bv;
          v = v > 0.0f ? v : 0.0f;
          p[(size_t)e * N] = (bf16)v;
        }
      } else {
        float* p = (float*)Cout + (size_t)m0 * N + n;
#pragma unroll
        for (int e = 0; e < 8; ++e)
          p[(size_t)e * N] = acc[mi][ni][e] + bv;
      }
    }
  }
}

// ===================== windowed graph attention + residual + BN1 =====================
// QKV: [B*17, 1536] (q|k|v), AKV: [B*34, 1024] (ak|av)
__global__ __launch_bounds__(256) void attn_kernel(
    const float* __restrict__ QKV, const float* __restrict__ AKV,
    const float* __restrict__ x,
    const float* __restrict__ g1, const float* __restrict__ bb1,
    const float* __restrict__ m1, const float* __restrict__ v1,
    float* __restrict__ retf, bf16* __restrict__ retbf)
{
  const int bl = blockIdx.x;
  const int b = bl / LJ, l = bl % LJ;
  const int h = threadIdx.x >> 5;   // wave = head
  const int lane = threadIdx.x & 31;
  const int slen = c_slen[l];

  const float* qrow = QKV + (size_t)bl * 1536 + h * DK;
  const float q0 = qrow[lane], q1 = qrow[lane + 32];

  float s[7];
  float mx = -__builtin_inff();
#pragma unroll
  for (int j = 0; j < 7; ++j) {
    bool valid; const float* krow = nullptr;
    if (j < 5) {
      valid = (j < slen);
      const int nl = valid ? c_neigh[l][j] : 0;
      krow = QKV + (size_t)(b * LJ + nl) * 1536 + 512 + h * DK;
    } else {
      valid = true;
      const int a = j - 5;
      krow = AKV + (size_t)(b * 2 * LJ + a * LJ + l) * 1024 + h * DK;
    }
    float p = valid ? (q0 * krow[lane] + q1 * krow[lane + 32]) : 0.0f;
#pragma unroll
    for (int off = 16; off; off >>= 1) p += __shfl_xor(p, off, 32);
    s[j] = valid ? p * 0.125f : -__builtin_inff();   // 1/sqrt(64)
    mx = s[j] > mx ? s[j] : mx;
  }
  float denom = 0.0f;
#pragma unroll
  for (int j = 0; j < 7; ++j) {
    s[j] = (s[j] == -__builtin_inff()) ? 0.0f : __expf(s[j] - mx);
    denom += s[j];
  }
  const float inv = 1.0f / denom;
  float o0 = 0.0f, o1 = 0.0f;
#pragma unroll
  for (int j = 0; j < 7; ++j) {
    if (s[j] == 0.0f) continue;   // uniform across the wave
    const float* vrow;
    if (j < 5) {
      vrow = QKV + (size_t)(b * LJ + c_neigh[l][j]) * 1536 + 1024 + h * DK;
    } else {
      vrow = AKV + (size_t)(b * 2 * LJ + (j - 5) * LJ + l) * 1024 + 512 + h * DK;
    }
    const float a = s[j] * inv;
    o0 += a * vrow[lane]; o1 += a * vrow[lane + 32];
  }
  // residual + BN1 for channels c0, c1
  const int c0 = h * DK + lane, c1 = c0 + 32;
  const float r0 = x[(size_t)b * NHID * LJ + (size_t)c0 * LJ + l] + o0;
  const float r1 = x[(size_t)b * NHID * LJ + (size_t)c1 * LJ + l] + o1;
  const float n0 = (r0 - m1[c0]) * g1[c0] * __frsqrt_rn(v1[c0] + 1e-5f) + bb1[c0];
  const float n1 = (r1 - m1[c1]) * g1[c1] * __frsqrt_rn(v1[c1] + 1e-5f) + bb1[c1];
  retf[(size_t)bl * NHID + c0] = n0;  retbf[(size_t)bl * NHID + c0] = (bf16)n0;
  retf[(size_t)bl * NHID + c1] = n1;  retbf[(size_t)bl * NHID + c1] = (bf16)n1;
}

// ===================== FFN2 epilogue: +b2 +residual, BN2, transpose to [B,C,L] =====================
__global__ void epilogue_kernel(
    const float* __restrict__ Z, const float* __restrict__ retf,
    const float* __restrict__ b2,
    const float* __restrict__ g2, const float* __restrict__ bb2,
    const float* __restrict__ m2, const float* __restrict__ v2,
    float* __restrict__ out)
{
  const int bl = blockIdx.x; const int b = bl / LJ, l = bl % LJ;
  for (int c = threadIdx.x; c < NHID; c += blockDim.x) {
    const float z = Z[(size_t)bl * NHID + c] + b2[c] + retf[(size_t)bl * NHID + c];
    const float n = (z - m2[c]) * g2[c] * __frsqrt_rn(v2[c] + 1e-5f) + bb2[c];
    out[(size_t)b * NHID * LJ + (size_t)c * LJ + l] = n;
  }
}

// ===================== host-side launch =====================
extern "C" void kernel_launch(void* const* d_in, const int* in_sizes, int n_in,
                              void* d_out, int out_size, void* d_ws, size_t ws_size,
                              hipStream_t stream) {
  const float* x    = (const float*)d_in[0];
  const float* ax   = (const float*)d_in[1];
  const float* Wq_w = (const float*)d_in[2];
  const float* Wq_b = (const float*)d_in[3];
  const float* Wk_w = (const float*)d_in[4];
  const float* Wk_b = (const float*)d_in[5];
  const float* Wv_w = (const float*)d_in[6];
  const float* Wv_b = (const float*)d_in[7];
  const float* bn1g = (const float*)d_in[8];
  const float* bn1b = (const float*)d_in[9];
  const float* bn1m = (const float*)d_in[10];
  const float* bn1v = (const float*)d_in[11];
  const float* W1   = (const float*)d_in[12];
  const float* b1   = (const float*)d_in[13];
  const float* W2   = (const float*)d_in[14];
  const float* b2   = (const float*)d_in[15];
  const float* bn2g = (const float*)d_in[16];
  const float* bn2b = (const float*)d_in[17];
  const float* bn2m = (const float*)d_in[18];
  const float* bn2v = (const float*)d_in[19];
  float* out = (float*)d_out;

  char* w = (char*)d_ws;
  size_t off = 0;
  auto carve = [&](size_t bytes) -> void* {
    void* p = w + off;
    off = (off + bytes + 255) & ~(size_t)255;
    return p;
  };
  bf16*  Xbf    = (bf16*)carve((size_t)M_X * NHID * 2);
  bf16*  XAbf   = (bf16*)carve((size_t)M_AX * NHID * 2);
  bf16*  Wqkvbf = (bf16*)carve((size_t)3 * NHID * NHID * 2);      // [1536,512]
  bf16*  W1bf   = (bf16*)carve((size_t)DFF * NHID * 2);           // [2048,512]
  bf16*  W2bf   = (bf16*)carve((size_t)NHID * DFF * 2);           // [512,2048]
  float* biascat= (float*)carve((size_t)3 * NHID * 4);
  float* QKV    = (float*)carve((size_t)M_X * 3 * NHID * 4);      // [4352,1536]
  float* AKV    = (float*)carve((size_t)M_AX * 2 * NHID * 4);     // [8704,1024]
  float* retf   = (float*)carve((size_t)M_X * NHID * 4);
  bf16*  retbf  = (bf16*)carve((size_t)M_X * NHID * 2);
  bf16*  Hbf    = (bf16*)carve((size_t)M_X * DFF * 2);
  float* Zf     = (float*)carve((size_t)M_X * NHID * 4);
  (void)ws_size; (void)n_in; (void)in_sizes; (void)out_size;

  const int WN = NHID * NHID; // 262144
  cvt_f32_bf16<<<(WN + 255) / 256, 256, 0, stream>>>(Wq_w, Wqkvbf, WN);
  cvt_f32_bf16<<<(WN + 255) / 256, 256, 0, stream>>>(Wk_w, Wqkvbf + WN, WN);
  cvt_f32_bf16<<<(WN + 255) / 256, 256, 0, stream>>>(Wv_w, Wqkvbf + 2 * WN, WN);
  cvt_f32_bf16<<<(DFF * NHID + 255) / 256, 256, 0, stream>>>(W1, W1bf, DFF * NHID);
  cvt_f32_bf16<<<(NHID * DFF + 255) / 256, 256, 0, stream>>>(W2, W2bf, NHID * DFF);
  concat_bias_kernel<<<6, 256, 0, stream>>>(Wq_b, Wk_b, Wv_b, biascat);
  cvt_x_kernel<<<M_X, 256, 0, stream>>>(x, Xbf);
  cvt_ax_kernel<<<M_AX, 256, 0, stream>>>(ax, XAbf);

  // Q|K|V projections from x: [4352,512] x [1536,512]^T
  gemm_bf16_wmma<0><<<dim3(1536 / BN, M_X / BM), 256, 0, stream>>>(
      Xbf, Wqkvbf, biascat, QKV, M_X, 1536, NHID);
  // aK|aV projections from ax: [8704,512] x [1024,512]^T (Wk;Wv rows)
  gemm_bf16_wmma<0><<<dim3(1024 / BN, M_AX / BM), 256, 0, stream>>>(
      XAbf, Wqkvbf + WN, biascat + NHID, AKV, M_AX, 1024, NHID);

  attn_kernel<<<M_X, 256, 0, stream>>>(QKV, AKV, x, bn1g, bn1b, bn1m, bn1v, retf, retbf);

  // FFN1: relu([4352,512] x [2048,512]^T + b1) -> bf16
  gemm_bf16_wmma<1><<<dim3(DFF / BN, M_X / BM), 256, 0, stream>>>(
      retbf, W1bf, b1, Hbf, M_X, DFF, NHID);
  // FFN2: [4352,2048] x [512,2048]^T -> f32 raw
  gemm_bf16_wmma<2><<<dim3(NHID / BN, M_X / BM), 256, 0, stream>>>(
      Hbf, W2bf, nullptr, Zf, M_X, NHID, DFF);

  epilogue_kernel<<<M_X, 256, 0, stream>>>(Zf, retf, b2, bn2g, bn2b, bn2m, bn2v, out);
}
